// DynamicTokenMixing_90915867722021
// MI455X (gfx1250) — matrix operand order repaired and checked by reading, not compiled, not run on hardware
//
#include <hip/hip_runtime.h>
#include <hip/hip_bf16.h>

// ---------------------------------------------------------------------------
// Types / helpers
// ---------------------------------------------------------------------------
typedef __bf16 bf16_t;
typedef __attribute__((ext_vector_type(16))) __bf16 v16bf;
typedef __attribute__((ext_vector_type(8)))  float  v8f;

union frag_t { v16bf v; uint4 q[2]; };

static __device__ __forceinline__ bf16_t f2bf(float f) {
    unsigned u = __float_as_uint(f);
    unsigned r = (u + 0x7FFFu + ((u >> 16) & 1u)) >> 16;   // RNE
    unsigned short s = (unsigned short)r;
    bf16_t b;
    __builtin_memcpy(&b, &s, 2);
    return b;
}

#define DIM     1024
#define TOK     4096
#define QKV_LD  3072   // [ Q(1024) | K(1024) | V(1024) ] per token, head-natural layout

// --------- CDNA5 async global->LDS copy (guarded; falls back to reg staging) --
#ifndef __has_builtin
#define __has_builtin(x) 0
#endif
#if __has_builtin(__builtin_amdgcn_global_load_async_to_lds_b128)
#define ASYNC_LDS 1
#else
#define ASYNC_LDS 0
#endif

#if ASYNC_LDS
typedef int v4i_t __attribute__((vector_size(16)));
typedef __attribute__((address_space(1))) v4i_t* gptr4_t;   // global AS1
typedef __attribute__((address_space(3))) v4i_t* lptr4_t;   // LDS AS3

static __device__ __forceinline__ void async_cp16(const bf16_t* g, bf16_t* l) {
    __builtin_amdgcn_global_load_async_to_lds_b128(
        (gptr4_t)(v4i_t*)g,      // addrspacecast generic -> AS1
        (lptr4_t)(v4i_t*)l,      // addrspacecast generic -> AS3
        0 /*imm offset*/, 0 /*cpol*/);
}
static __device__ __forceinline__ void wait_async0() {
    asm volatile("s_wait_asynccnt 0x0" ::: "memory");
}
#endif

// ---------------------------------------------------------------------------
// Kernel 1a: x (f32) -> x_bf (bf16), 4 elems / thread
// ---------------------------------------------------------------------------
__global__ void k_convert_x(const float* __restrict__ x, bf16_t* __restrict__ xb) {
    size_t i = (size_t)blockIdx.x * blockDim.x + threadIdx.x;   // element-quad index
    float4 v = ((const float4*)x)[i];
    union { bf16_t b[4]; uint2 u; } pk;
    pk.b[0] = f2bf(v.x); pk.b[1] = f2bf(v.y);
    pk.b[2] = f2bf(v.z); pk.b[3] = f2bf(v.w);
    ((uint2*)xb)[i] = pk.u;
}

// ---------------------------------------------------------------------------
// Kernel 1b: build Wqkv^T (3072 x 1024 bf16) with GQA interleave folded in.
// ---------------------------------------------------------------------------
__global__ void k_pack_wqkvT(const float* __restrict__ Wq,
                             const float* __restrict__ Wkv,
                             bf16_t* __restrict__ WT) {
    size_t idx = (size_t)blockIdx.x * blockDim.x + threadIdx.x;  // n*1024 + k
    int k = (int)(idx & (DIM - 1));
    int n = (int)(idx >> 10);
    float src;
    if (n < DIM) {
        src = Wq[(size_t)k * DIM + n];
    } else if (n < 2 * DIM) {
        int hh = (n - DIM) >> 6, dd = (n - DIM) & 63;
        int col = (hh < 8) ? (hh * 64 + dd) : (1024 + (hh - 8) * 64 + dd);
        src = Wkv[(size_t)k * (2 * DIM) + col];
    } else {
        int hh = (n - 2 * DIM) >> 6, dd = (n - 2 * DIM) & 63;
        int col = (hh < 8) ? (512 + hh * 64 + dd) : (1536 + (hh - 8) * 64 + dd);
        src = Wkv[(size_t)k * (2 * DIM) + col];
    }
    WT[idx] = f2bf(src);
}

// Kernel 1c: Wproj^T (1024 x 1024 bf16)
__global__ void k_pack_wpT(const float* __restrict__ W, bf16_t* __restrict__ WT) {
    size_t idx = (size_t)blockIdx.x * blockDim.x + threadIdx.x;  // n*1024 + k
    int k = (int)(idx & (DIM - 1));
    int n = (int)(idx >> 10);
    WT[idx] = f2bf(W[(size_t)k * DIM + n]);
}

// ---------------------------------------------------------------------------
// Kernel 2/4: LDS-tiled, double-buffered bf16 WMMA GEMM.
//   C[M,N] = A[M,K] * BT[N,K]^T (+bias)
//   Block tile: 256(M) x 64(N), k-step 32. 8 waves; wave w -> rows 32w..32w+31.
//   8 WMMAs / wave / k-step; B tile shared by all 8 waves (8x reuse).
// ---------------------------------------------------------------------------
#define KSTEP 32
#define MT    256
#define NT    64

__global__ void __launch_bounds__(256)
k_gemm_wmma(const bf16_t* __restrict__ A,
            const bf16_t* __restrict__ BT,
            void* __restrict__ Cout,
            const float* __restrict__ bias,
            int M, int N, int K, int out_is_f32) {
    __shared__ __align__(16) bf16_t ldsA[2][MT * KSTEP];   // [row][k] 16 KB x2
    __shared__ __align__(16) bf16_t ldsB[2][NT * KSTEP];   // [n][k]    4 KB x2

    const int tid  = threadIdx.x;
    const int lane = tid & 31;
    const int wave = tid >> 5;
    const int half = lane >> 4;
    const int l15  = lane & 15;

    const int ntn = N >> 6;
    const int bm  = blockIdx.x / ntn;
    const int bn  = blockIdx.x % ntn;
    const int m0  = bm * MT;
    const int n0  = bn * NT;

    // staging geometry: thread owns 4 A chunks (rows r0+64c) + 1 B chunk, 16B each
    const int ko = (tid & 3) * 8;    // k offset of this thread's chunk
    const int r0 = tid >> 2;         // 0..63
    const bf16_t* aBase = A  + (size_t)(m0 + r0) * K + ko;
    const bf16_t* bBase = BT + (size_t)(n0 + r0) * K + ko;

    const int nsteps = K / KSTEP;

    v8f zero = {0.f, 0.f, 0.f, 0.f, 0.f, 0.f, 0.f, 0.f};
    v8f acc0[4] = {zero, zero, zero, zero};   // rows wave*32 .. +15
    v8f acc1[4] = {zero, zero, zero, zero};   // rows wave*32+16 .. +31

#if ASYNC_LDS
    // ---- prologue: async stage tile 0 into buffer 0 ----
#pragma unroll
    for (int c = 0; c < 4; ++c)
        async_cp16(aBase + (size_t)c * 64 * K, &ldsA[0][(r0 + c * 64) * KSTEP + ko]);
    async_cp16(bBase, &ldsB[0][r0 * KSTEP + ko]);
    wait_async0();
    __syncthreads();
#else
    uint4 ra[4], rb;
#pragma unroll
    for (int c = 0; c < 4; ++c)
        ra[c] = *(const uint4*)(aBase + (size_t)c * 64 * K);
    rb = *(const uint4*)(bBase);
#pragma unroll
    for (int c = 0; c < 4; ++c)
        *(uint4*)(&ldsA[0][(r0 + c * 64) * KSTEP + ko]) = ra[c];
    *(uint4*)(&ldsB[0][r0 * KSTEP + ko]) = rb;
    __syncthreads();
#endif

    for (int ks = 0; ks < nsteps; ++ks) {
        const int cur = ks & 1;
        const int k0n = (ks + 1) * KSTEP;
        const bool more = (ks + 1 < nsteps);

#if ASYNC_LDS
        if (more) {   // stage next tile into the other buffer (last read in the previous iteration)
#pragma unroll
            for (int c = 0; c < 4; ++c)
                async_cp16(aBase + (size_t)c * 64 * K + k0n,
                           &ldsA[cur ^ 1][(r0 + c * 64) * KSTEP + ko]);
            async_cp16(bBase + k0n, &ldsB[cur ^ 1][r0 * KSTEP + ko]);
        }
#else
        if (more) {   // global -> regs; consumed after compute
#pragma unroll
            for (int c = 0; c < 4; ++c)
                ra[c] = *(const uint4*)(aBase + (size_t)c * 64 * K + k0n);
            rb = *(const uint4*)(bBase + k0n);
        }
#endif

        // ---- compute from LDS buffer `cur` ----
        frag_t a0, a1;
        {
            const bf16_t* la = &ldsA[cur][(wave * 32 + l15) * KSTEP];
            const int kb = half * 8;
            a0.q[0] = *(const uint4*)(la + kb);
            a0.q[1] = *(const uint4*)(la + kb + 16);
            la += 16 * KSTEP;
            a1.q[0] = *(const uint4*)(la + kb);
            a1.q[1] = *(const uint4*)(la + kb + 16);
        }
#pragma unroll
        for (int t = 0; t < 4; ++t) {
            frag_t b;
            const bf16_t* lb = &ldsB[cur][(t * 16 + l15) * KSTEP + half * 8];
            b.q[0] = ((const uint4*)lb)[0];
            b.q[1] = ((const uint4*)lb)[2];
            acc0[t] = __builtin_amdgcn_wmma_f32_16x16x32_bf16(
                          false, a0.v, false, b.v, (short)0, acc0[t], false, false);
            acc1[t] = __builtin_amdgcn_wmma_f32_16x16x32_bf16(
                          false, a1.v, false, b.v, (short)0, acc1[t], false, false);
        }

#if ASYNC_LDS
        if (more) wait_async0();
#else
        if (more) {
#pragma unroll
            for (int c = 0; c < 4; ++c)
                *(uint4*)(&ldsA[cur ^ 1][(r0 + c * 64) * KSTEP + ko]) = ra[c];
            *(uint4*)(&ldsB[cur ^ 1][r0 * KSTEP + ko]) = rb;
        }
#endif
        __syncthreads();
    }

    // ---- epilogue ----
#pragma unroll
    for (int f = 0; f < 2; ++f) {
#pragma unroll
        for (int t = 0; t < 4; ++t) {
            const int n = n0 + t * 16 + l15;
            const v8f* accp = f ? &acc1[t] : &acc0[t];
#pragma unroll
            for (int r = 0; r < 8; ++r) {
                const int row = m0 + wave * 32 + f * 16 + half * 8 + r;
                if (out_is_f32) {
                    float bsum = bias ? bias[n] : 0.f;
                    ((float*)Cout)[(size_t)row * N + n] = (*accp)[r] + bsum;
                } else {
                    ((bf16_t*)Cout)[(size_t)row * N + n] = f2bf((*accp)[r]);
                }
            }
        }
    }
}

// ---------------------------------------------------------------------------
// Kernel 3: flash attention.  grid = (32 qblocks, 16 heads), block = 256.
// Each wave owns 16 query rows; K/V tiles (32 keys) staged in LDS per block.
// Logit scale = g^2 * d^-0.5 = 0.5.
// ---------------------------------------------------------------------------
__global__ void __launch_bounds__(256)
k_flash_attn(const bf16_t* __restrict__ QKV, bf16_t* __restrict__ Obf) {
    __shared__ __align__(16) bf16_t ldsK[32 * 64];      // [key][d]
    __shared__ __align__(16) bf16_t ldsVT[64 * 32];     // [d][key]
    __shared__ __align__(16) bf16_t ldsP[8][16 * 32];   // per-wave P scratch

    const int lane = threadIdx.x & 31;
    const int wave = threadIdx.x >> 5;
    const int half = lane >> 4;
    const int l15  = lane & 15;
    const int h    = blockIdx.y;
    const int q0   = blockIdx.x * 128 + wave * 16;

    // Q fragments (16 x 64, two K=32 halves)
    frag_t aq0, aq1;
    {
        const bf16_t* qrow = QKV + (size_t)(q0 + l15) * QKV_LD + h * 64;
        const int kb = half * 8;
        aq0.q[0] = *(const uint4*)(qrow + kb);
        aq0.q[1] = *(const uint4*)(qrow + kb + 16);
        aq1.q[0] = *(const uint4*)(qrow + 32 + kb);
        aq1.q[1] = *(const uint4*)(qrow + 32 + kb + 16);
    }

    v8f zero = {0.f, 0.f, 0.f, 0.f, 0.f, 0.f, 0.f, 0.f};
    v8f o[4] = {zero, zero, zero, zero};
    float mrow[8], lrow[8];
#pragma unroll
    for (int r = 0; r < 8; ++r) { mrow[r] = -1e30f; lrow[r] = 0.f; }

    for (int kt = 0; kt < TOK; kt += 32) {
        __syncthreads();
        {   // cooperative staging: 32 keys x 64 d, 8 elems / thread
            const int row = threadIdx.x >> 3;
            const int c8  = (threadIdx.x & 7) * 8;
            const bf16_t* ksrc = QKV + (size_t)(kt + row) * QKV_LD + 1024 + h * 64 + c8;
            *(uint4*)(&ldsK[row * 64 + c8]) = *(const uint4*)ksrc;
            const bf16_t* vsrc = QKV + (size_t)(kt + row) * QKV_LD + 2048 + h * 64 + c8;
#pragma unroll
            for (int e = 0; e < 8; ++e)
                ldsVT[(c8 + e) * 32 + row] = vsrc[e];
        }
        __syncthreads();

        // S = Q * K^T : keys 0..15 -> s0, 16..31 -> s1
        v8f s0 = zero, s1 = zero;
        const int ddb = half * 8;
        {
            frag_t b0, b1;
            b0.q[0] = ((const uint4*)(&ldsK[l15 * 64 + ddb]))[0];
            b0.q[1] = ((const uint4*)(&ldsK[l15 * 64 + ddb]))[2];
            b1.q[0] = ((const uint4*)(&ldsK[(16 + l15) * 64 + ddb]))[0];
            b1.q[1] = ((const uint4*)(&ldsK[(16 + l15) * 64 + ddb]))[2];
            s0 = __builtin_amdgcn_wmma_f32_16x16x32_bf16(false, aq0.v, false, b0.v, (short)0, s0, false, false);
            s1 = __builtin_amdgcn_wmma_f32_16x16x32_bf16(false, aq0.v, false, b1.v, (short)0, s1, false, false);
        }
        {
            frag_t b0, b1;
            b0.q[0] = ((const uint4*)(&ldsK[l15 * 64 + 32 + ddb]))[0];
            b0.q[1] = ((const uint4*)(&ldsK[l15 * 64 + 32 + ddb]))[2];
            b1.q[0] = ((const uint4*)(&ldsK[(16 + l15) * 64 + 32 + ddb]))[0];
            b1.q[1] = ((const uint4*)(&ldsK[(16 + l15) * 64 + 32 + ddb]))[2];
            s0 = __builtin_amdgcn_wmma_f32_16x16x32_bf16(false, aq1.v, false, b0.v, (short)0, s0, false, false);
            s1 = __builtin_amdgcn_wmma_f32_16x16x32_bf16(false, aq1.v, false, b1.v, (short)0, s1, false, false);
        }

        // online softmax (C layout: lane = col, VGPR = row)
#pragma unroll
        for (int r = 0; r < 8; ++r) {
            float v0 = s0[r] * 0.5f;
            float v1 = s1[r] * 0.5f;
            float mx = fmaxf(v0, v1);
            for (int msk = 1; msk < 16; msk <<= 1)
                mx = fmaxf(mx, __shfl_xor(mx, msk, 32));
            float mnew  = fmaxf(mrow[r], mx);
            float p0    = __expf(v0 - mnew);
            float p1    = __expf(v1 - mnew);
            float ps    = p0 + p1;
            for (int msk = 1; msk < 16; msk <<= 1)
                ps += __shfl_xor(ps, msk, 32);
            float alpha = __expf(mrow[r] - mnew);
            lrow[r] = lrow[r] * alpha + ps;
            mrow[r] = mnew;
            o[0][r] *= alpha; o[1][r] *= alpha; o[2][r] *= alpha; o[3][r] *= alpha;
            const int prow = half * 8 + r;
            ldsP[wave][prow * 32 + l15]      = f2bf(p0);
            ldsP[wave][prow * 32 + 16 + l15] = f2bf(p1);
        }

        // reload P as A fragment (16 x 32)
        frag_t ap;
        {
            const int kb = half * 8;
            ap.q[0] = *(const uint4*)(&ldsP[wave][l15 * 32 + kb]);
            ap.q[1] = *(const uint4*)(&ldsP[wave][l15 * 32 + kb + 16]);
        }

        // O += P * V
        const int kh = half * 8;
#pragma unroll
        for (int t = 0; t < 4; ++t) {
            frag_t bv;
            const int dcol = t * 16 + l15;
            bv.q[0] = ((const uint4*)(&ldsVT[dcol * 32 + kh]))[0];
            bv.q[1] = ((const uint4*)(&ldsVT[dcol * 32 + kh]))[2];
            o[t] = __builtin_amdgcn_wmma_f32_16x16x32_bf16(
                       false, ap.v, false, bv.v, (short)0, o[t], false, false);
        }
    }

    // normalize + store (4096 x 1024 bf16)
#pragma unroll
    for (int r = 0; r < 8; ++r) {
        float inv = 1.0f / lrow[r];
        const int row = q0 + half * 8 + r;
#pragma unroll
        for (int t = 0; t < 4; ++t) {
            const int col = h * 64 + t * 16 + l15;
            Obf[(size_t)row * DIM + col] = f2bf(o[t][r] * inv);
        }
    }
}

// ---------------------------------------------------------------------------
// Host launch
// ---------------------------------------------------------------------------
extern "C" void kernel_launch(void* const* d_in, const int* in_sizes, int n_in,
                              void* d_out, int out_size, void* d_ws, size_t ws_size,
                              hipStream_t stream) {
    const float* x     = (const float*)d_in[0];
    const float* Wq    = (const float*)d_in[1];
    const float* Wkv   = (const float*)d_in[2];
    const float* Wproj = (const float*)d_in[3];
    const float* bproj = (const float*)d_in[4];
    float* out = (float*)d_out;

    char* ws = (char*)d_ws;
    bf16_t* x_bf  = (bf16_t*)(ws);                       //  8 MB: 4096x1024
    bf16_t* wqkvT = (bf16_t*)(ws + 8388608);             //  6 MB: 3072x1024
    bf16_t* wpT   = (bf16_t*)(ws + 14680064);            //  2 MB: 1024x1024
    bf16_t* qkv   = (bf16_t*)(ws + 16777216);            // 24 MB: 4096x3072
    bf16_t* attn  = (bf16_t*)(ws + 41943040);            //  8 MB: 4096x1024

    // 1) conversions / weight packing
    k_convert_x <<<(TOK * DIM / 4) / 256, 256, 0, stream>>>(x, x_bf);
    k_pack_wqkvT<<<(3072 * DIM) / 256, 256, 0, stream>>>(Wq, Wkv, wqkvT);
    k_pack_wpT  <<<(DIM * DIM) / 256, 256, 0, stream>>>(Wproj, wpT);

    // 2) QKV = x @ Wqkv (4096 x 3072, bf16): (4096/256)*(3072/64) = 768 blocks
    k_gemm_wmma<<<768, 256, 0, stream>>>(x_bf, wqkvT, (void*)qkv, nullptr,
                                         TOK, 3072, DIM, 0);

    // 3) attention (16 heads, 128 queries / block)
    k_flash_attn<<<dim3(TOK / 128, 16), 256, 0, stream>>>(qkv, attn);

    // 4) out = attn @ Wproj + bproj (f32): (4096/256)*(1024/64) = 256 blocks
    k_gemm_wmma<<<256, 256, 0, stream>>>(attn, wpT, (void*)out, bproj,
                                         TOK, DIM, DIM, 1);
}
